// SplitterModel_66340064854429
// MI455X (gfx1250) — matrix operand order repaired and checked by reading, not compiled, not run on hardware
//
#include <hip/hip_runtime.h>
#include <math.h>

typedef __bf16 bf16;
typedef __attribute__((ext_vector_type(16))) __bf16 v16bf;
typedef __attribute__((ext_vector_type(8)))  __bf16 v8bf;
typedef __attribute__((ext_vector_type(8)))  float  v8f;

#define SEQ    2048
#define DM     768
#define NHEAD  12
#define DHEAD  64
#define NCH    16
#define CHK    128
#define BAND   384
#define FFD    3072
#define NLAYER 12
#define NEGV   (-1e9f)
#define LDSB_STRIDE 40   // GEMM weight tile: padded col stride (elements)
#define K3_STRIDE   72   // scores k3 tile: padded row stride (elements)
#define VT_STRIDE   392  // attn_out transposed V tile: padded col stride

// ---------------------------------------------------------------- helpers

__device__ __forceinline__ v8f wmma_bf16(v16bf a, v16bf b, v8f c) {
  return __builtin_amdgcn_wmma_f32_16x16x32_bf16(false, a, false, b, (short)0, c,
                                                 false, false);
}

__device__ __forceinline__ v16bf combine8(v8bf lo, v8bf hi) {
  v16bf f;
#pragma unroll
  for (int i = 0; i < 8; ++i) { f[i] = lo[i]; f[8 + i] = hi[i]; }
  return f;
}

// A fragment: 16x32 (MxK) bf16, row-major source, p = row base + kBase.
// lanes 0-15: K 0..15; lanes 16-31: K 16..31. Two 16B loads.
__device__ __forceinline__ v16bf load_a_frag(const bf16* p, int hsel) {
  const bf16* q = p + hsel * 16;
  return combine8(*(const v8bf*)q, *(const v8bf*)(q + 8));
}

// B fragment: 32x16 (KxN) bf16 where K is contiguous in memory per column.
// p = column base + kBase. lanes 0-15: K 0..15; lanes 16-31: K 16..31.
__device__ __forceinline__ v16bf load_b_contig(const bf16* p, int hsel) {
  const bf16* q = p + hsel * 16;
  return combine8(*(const v8bf*)q, *(const v8bf*)(q + 8));
}

// B fragment from LDS weight tile stored transposed [col][LDSB_STRIDE].
__device__ __forceinline__ v16bf load_b_lds(const bf16* ldsW, int col, int hsel) {
  const bf16* p = ldsW + col * LDSB_STRIDE + hsel * 16;
  return combine8(*(const v8bf*)p, *(const v8bf*)(p + 8));
}

__device__ __forceinline__ float gelu_f(float v) {
  return 0.5f * v * (1.0f + erff(v * 0.70710678118654752f));
}

__device__ __forceinline__ float block_sum256(float v, float* red) {
  int tid = threadIdx.x;
  red[tid] = v; __syncthreads();
#pragma unroll
  for (int off = 128; off > 0; off >>= 1) {
    if (tid < off) red[tid] += red[tid + off];
    __syncthreads();
  }
  float r = red[0]; __syncthreads();
  return r;
}

__device__ __forceinline__ float block_max256(float v, float* red) {
  int tid = threadIdx.x;
  red[tid] = v; __syncthreads();
#pragma unroll
  for (int off = 128; off > 0; off >>= 1) {
    if (tid < off) red[tid] = fmaxf(red[tid], red[tid + off]);
    __syncthreads();
  }
  float r = red[0]; __syncthreads();
  return r;
}

// ---------------------------------------------------------------- embedding + LN

__global__ void embed_ln_kernel(const int* __restrict__ ids,
                                const float* __restrict__ wemb,
                                const float* __restrict__ pemb,
                                const float* __restrict__ g,
                                const float* __restrict__ b,
                                float* __restrict__ x, bf16* __restrict__ xb) {
  __shared__ float red[256];
  int s = blockIdx.x, tid = threadIdx.x;
  size_t wb = (size_t)ids[s] * DM;
  size_t pbase = (size_t)s * DM;
  float e[3]; float ls = 0.f;
#pragma unroll
  for (int i = 0; i < 3; ++i) {
    int d = tid + i * 256;
    e[i] = wemb[wb + d] + pemb[pbase + d];
    ls += e[i];
  }
  float mean = block_sum256(ls, red) * (1.0f / DM);
  float lv = 0.f;
#pragma unroll
  for (int i = 0; i < 3; ++i) { float t = e[i] - mean; lv += t * t; }
  float rstd = rsqrtf(block_sum256(lv, red) * (1.0f / DM) + 1e-5f);
#pragma unroll
  for (int i = 0; i < 3; ++i) {
    int d = tid + i * 256;
    float y = (e[i] - mean) * rstd * g[d] + b[d];
    x[pbase + d] = y;
    xb[pbase + d] = (bf16)y;
  }
}

// y = LN(x + r); x updated in place, bf16 copy written.
__global__ void add_ln_kernel(float* __restrict__ x, const float* __restrict__ r,
                              const float* __restrict__ g, const float* __restrict__ b,
                              bf16* __restrict__ xb) {
  __shared__ float red[256];
  int s = blockIdx.x, tid = threadIdx.x;
  size_t base = (size_t)s * DM;
  float e[3]; float ls = 0.f;
#pragma unroll
  for (int i = 0; i < 3; ++i) {
    int d = tid + i * 256;
    e[i] = x[base + d] + r[base + d];
    ls += e[i];
  }
  float mean = block_sum256(ls, red) * (1.0f / DM);
  float lv = 0.f;
#pragma unroll
  for (int i = 0; i < 3; ++i) { float t = e[i] - mean; lv += t * t; }
  float rstd = rsqrtf(block_sum256(lv, red) * (1.0f / DM) + 1e-5f);
#pragma unroll
  for (int i = 0; i < 3; ++i) {
    int d = tid + i * 256;
    float y = (e[i] - mean) * rstd * g[d] + b[d];
    x[base + d] = y;
    xb[base + d] = (bf16)y;
  }
}

// ---------------------------------------------------------------- WMMA GEMM
// C[M,N] = A(bf16,row-major,MxK) @ W(f32,row-major,KxN) + bias.
// Block: 256 threads = 8 waves; block tile 128x64; wave tile 32x32.
// Weight tile staged in LDS transposed ([col][k], padded stride) as bf16 so
// B fragments are two contiguous ds_load_b128 per lane.

__global__ void gemm_wmma_kernel(const bf16* __restrict__ A,
                                 const float* __restrict__ W,
                                 const float* __restrict__ bias,
                                 float* __restrict__ Cf, bf16* __restrict__ Cbf,
                                 int M, int N, int K, int act, float outScale) {
  __shared__ bf16 ldsW[64 * LDSB_STRIDE];
  int tid = threadIdx.x;
  int wave = tid >> 5, lane = tid & 31;
  int waveM = wave & 3, waveN = wave >> 2;
  int m16 = lane & 15, hsel = lane >> 4;
  int rowBlock = blockIdx.y * 128;
  int colBlock = blockIdx.x * 64;
  int rBase = rowBlock + waveM * 32;   // global row base of wave tile
  int cBase = waveN * 32;              // col base within block tile

  const bf16* arow0 = A + (size_t)(rBase + m16) * K;
  const bf16* arow1 = A + (size_t)(rBase + 16 + m16) * K;

  // staging coords: this thread converts 8 K-values of one column per K-step
  int sc = tid & 63, sg = tid >> 6;    // column 0..63, k-group 0..3

  v8f acc00 = {}, acc01 = {}, acc10 = {}, acc11 = {};

  for (int kb = 0; kb < K; kb += 32) {
    __syncthreads();
    {
      const float* wp = W + (size_t)(kb + sg * 8) * N + colBlock + sc;
      if (kb + 32 < K) __builtin_prefetch(wp + (size_t)32 * N, 0, 1);
      v8bf tv;
#pragma unroll
      for (int j = 0; j < 8; ++j) tv[j] = (bf16)wp[(size_t)j * N];
      *(v8bf*)(ldsW + sc * LDSB_STRIDE + sg * 8) = tv;   // one ds_store_b128
    }
    __syncthreads();

    v16bf a0 = load_a_frag(arow0 + kb, hsel);
    v16bf a1 = load_a_frag(arow1 + kb, hsel);
    v16bf b0 = load_b_lds(ldsW, cBase + m16, hsel);
    v16bf b1 = load_b_lds(ldsW, cBase + 16 + m16, hsel);

    acc00 = wmma_bf16(a0, b0, acc00);
    acc01 = wmma_bf16(a0, b1, acc01);
    acc10 = wmma_bf16(a1, b0, acc10);
    acc11 = wmma_bf16(a1, b1, acc11);
  }

#pragma unroll
  for (int mt = 0; mt < 2; ++mt) {
#pragma unroll
    for (int nt = 0; nt < 2; ++nt) {
      v8f acc = (mt == 0) ? (nt == 0 ? acc00 : acc01) : (nt == 0 ? acc10 : acc11);
      int col = colBlock + cBase + nt * 16 + m16;
      float bs = bias[col];
#pragma unroll
      for (int r = 0; r < 8; ++r) {
        int row = rBase + mt * 16 + hsel * 8 + r;
        float v = acc[r] + bs;
        if (act == 1) v = gelu_f(v);
        if (Cf)  Cf[(size_t)row * N + col] = v;
        if (Cbf) Cbf[(size_t)row * N + col] = (bf16)(v * outScale);
      }
    }
  }
}

// ---------------------------------------------------------------- attention

// sg[h][s] = q[s,h]·k[0,h], masked by am[0]
__global__ void global_col_kernel(const bf16* __restrict__ qb,
                                  const bf16* __restrict__ kb,
                                  const int* __restrict__ am,
                                  float* __restrict__ sg) {
  int idx = blockIdx.x * 256 + threadIdx.x;   // NHEAD*SEQ total
  int h = idx >> 11, s = idx & 2047;
  const bf16* qp = qb + (size_t)s * DM + h * DHEAD;
  const bf16* kp = kb + h * DHEAD;            // key row 0
  float acc = 0.f;
#pragma unroll
  for (int d = 0; d < DHEAD; ++d) acc += (float)qp[d] * (float)kp[d];
  sg[h * SEQ + s] = (am[0] > 0) ? acc : NEGV;
}

// banded scores: per (chunk n, head h) block; scores[128,384] via WMMA + mask.
// k3 band tile staged into LDS row-major [j][K3_STRIDE] with CDNA5 async
// global->LDS copies (ASYNCcnt); B frags are then contiguous ds_load_b128.
__global__ void attn_scores_kernel(const bf16* __restrict__ qb,
                                   const bf16* __restrict__ kmat,
                                   const int* __restrict__ am,
                                   float* __restrict__ scores) {
  __shared__ bf16 ldsk[BAND * K3_STRIDE];     // ~54KB
  int n = blockIdx.x, h = blockIdx.y;
  int tid = threadIdx.x;
  int wave = tid >> 5, lane = tid & 31;       // wave = M-tile (0..7)
  int m16 = lane & 15, hsel = lane >> 4;

  // stage: 3072 chunks of 16B; valid rows via async copy, OOB rows zero-filled
  for (int i = tid; i < BAND * 8; i += 256) {
    int j = i >> 3, c = i & 7;                // band row, 16B chunk within row
    int krow = n * CHK - CHK + j;
    bf16* dst = ldsk + (size_t)j * K3_STRIDE + c * 8;
    if (krow >= 0 && krow < SEQ) {
      unsigned lds_off = (unsigned)(size_t)dst;
      unsigned long long ga =
          (unsigned long long)(size_t)(kmat + (size_t)krow * DM + h * DHEAD + c * 8);
      asm volatile("global_load_async_to_lds_b128 %0, %1, off"
                   :: "v"(lds_off), "v"(ga) : "memory");
    } else {
      v8bf z = {};
      *(v8bf*)dst = z;
    }
  }
  asm volatile("s_wait_asynccnt 0x0" ::: "memory");
  __syncthreads();

  const bf16* qrow = qb + (size_t)(n * CHK + wave * 16 + m16) * DM + h * DHEAD;
  v16bf aK0 = load_a_frag(qrow, hsel);        // K 0..31
  v16bf aK1 = load_a_frag(qrow + 32, hsel);   // K 32..63

  for (int nt = 0; nt < 24; ++nt) {
    int col = nt * 16 + m16;                  // band index j of this lane's column
    const bf16* kp = ldsk + (size_t)col * K3_STRIDE;
    v16bf bK0 = load_b_contig(kp, hsel);
    v16bf bK1 = load_b_contig(kp + 32, hsel);

    v8f acc = {};
    acc = wmma_bf16(aK0, bK0, acc);
    acc = wmma_bf16(aK1, bK1, acc);

#pragma unroll
    for (int r = 0; r < 8; ++r) {
      int ql = wave * 16 + hsel * 8 + r;      // 0..127
      int qi = n * CHK + ql;
      int j = nt * 16 + m16;
      int kj = n * CHK - CHK + j;
      int dd = qi - kj; if (dd < 0) dd = -dd;
      bool ok = (kj >= 0) && (kj < SEQ) && (kj != 0) && (dd <= CHK) && (am[kj] > 0);
      scores[((size_t)(h * NCH + n) * CHK + ql) * BAND + j] = ok ? acc[r] : NEGV;
    }
  }
}

// warp-per-row softmax over [sg | 384 band scores]; writes pg (f32) + pb (bf16)
__global__ void attn_softmax_kernel(const float* __restrict__ scores,
                                    const float* __restrict__ sg,
                                    float* __restrict__ pg, bf16* __restrict__ pb) {
  int row = blockIdx.x * 8 + (threadIdx.x >> 5);  // 0..NHEAD*SEQ-1
  int lane = threadIdx.x & 31;
  const float* srow = scores + (size_t)row * BAND;
  float sgv = sg[row];
  float mx = sgv;
  for (int j = lane; j < BAND; j += 32) mx = fmaxf(mx, srow[j]);
#pragma unroll
  for (int off = 16; off > 0; off >>= 1) mx = fmaxf(mx, __shfl_xor(mx, off, 32));
  float ls = 0.f;
  for (int j = lane; j < BAND; j += 32) ls += expf(srow[j] - mx);
#pragma unroll
  for (int off = 16; off > 0; off >>= 1) ls += __shfl_xor(ls, off, 32);
  float esg = expf(sgv - mx);
  float inv = 1.0f / (ls + esg);
  if (lane == 0) pg[row] = esg * inv;
  bf16* prow = pb + (size_t)row * BAND;
  for (int j = lane; j < BAND; j += 32)
    prow[j] = (bf16)(expf(srow[j] - mx) * inv);
}

// out[128,64] = pb_chunk @ v3 + pg * v[0]; writes bf16 att buffer.
// V band tile staged TRANSPOSED in LDS ([d][k], padded stride) so B fragments
// are two contiguous ds_load_b128; staging reads stay coalesced b128.
__global__ void attn_out_kernel(const bf16* __restrict__ pb,
                                const bf16* __restrict__ vb,
                                const float* __restrict__ pg,
                                bf16* __restrict__ attb) {
  __shared__ bf16 ldsvT[DHEAD * VT_STRIDE];   // ~49KB
  int n = blockIdx.x, h = blockIdx.y;
  int tid = threadIdx.x;
  int wave = tid >> 5, lane = tid & 31;       // wave = M-tile
  int m16 = lane & 15, hsel = lane >> 4;

  for (int i = tid; i < BAND * 8; i += 256) {
    int j = i >> 3, c = i & 7;                // band row, d-chunk (8 dims)
    int krow = n * CHK - CHK + j;
    v8bf val = {};
    if (krow >= 0 && krow < SEQ)
      val = *(const v8bf*)(vb + (size_t)krow * DM + h * DHEAD + c * 8);
#pragma unroll
    for (int t = 0; t < 8; ++t)
      ldsvT[(size_t)(c * 8 + t) * VT_STRIDE + j] = val[t];
  }
  __syncthreads();

  const bf16* arow = pb + ((size_t)(h * NCH + n) * CHK + wave * 16 + m16) * BAND;
  v8f acc[4] = {{}, {}, {}, {}};
  for (int kb = 0; kb < BAND; kb += 32) {
    v16bf a = load_a_frag(arow + kb, hsel);
#pragma unroll
    for (int nt = 0; nt < 4; ++nt) {
      const bf16* p = ldsvT + (size_t)(nt * 16 + m16) * VT_STRIDE + kb;
      v16bf b = load_b_contig(p, hsel);
      acc[nt] = wmma_bf16(a, b, acc[nt]);
    }
  }

#pragma unroll
  for (int nt = 0; nt < 4; ++nt) {
    int d = nt * 16 + m16;
    float v0 = (float)vb[h * DHEAD + d];      // value of token 0, this head
#pragma unroll
    for (int r = 0; r < 8; ++r) {
      int ml = wave * 16 + hsel * 8 + r;
      int s = n * CHK + ml;
      float val = acc[nt][r] + pg[h * SEQ + s] * v0;
      attb[(size_t)s * DM + h * DHEAD + d] = (bf16)val;
    }
  }
}

// qg0[j] = (x[0]·Wqg[:,j] + bqg[j]) * scale   (768 outputs)
__global__ void matvec_row0_kernel(const float* __restrict__ x,
                                   const float* __restrict__ Wg,
                                   const float* __restrict__ bg,
                                   float* __restrict__ qg0, float scale) {
  int j = blockIdx.x * 256 + threadIdx.x;
  float acc = bg[j];
  for (int k = 0; k < DM; ++k) acc += x[k] * Wg[(size_t)k * DM + j];
  qg0[j] = acc * scale;
}

// CLS global attention row: att[0, h*64+d] = softmax(qg0·kg) @ vg
__global__ void global_row_kernel(const float* __restrict__ qg0,
                                  const float* __restrict__ kg,
                                  const float* __restrict__ vg,
                                  const int* __restrict__ am,
                                  bf16* __restrict__ attb) {
  __shared__ float sprob[SEQ];
  __shared__ float red[256];
  __shared__ float part[4][DHEAD];
  int h = blockIdx.x, tid = threadIdx.x;
  const float* q = qg0 + h * DHEAD;

  float lmax = -3.0e38f;
  for (int s = tid; s < SEQ; s += 256) {
    const float* kp = kg + (size_t)s * DM + h * DHEAD;
    float acc = 0.f;
#pragma unroll
    for (int d = 0; d < DHEAD; ++d) acc += q[d] * kp[d];
    acc = (am[s] > 0) ? acc : NEGV;
    sprob[s] = acc;
    lmax = fmaxf(lmax, acc);
  }
  float mx = block_max256(lmax, red);
  float lsum = 0.f;
  for (int s = tid; s < SEQ; s += 256) {
    float e = expf(sprob[s] - mx);
    sprob[s] = e;
    lsum += e;
  }
  float tot = block_sum256(lsum, red);
  __syncthreads();

  int d = tid & 63, g = tid >> 6;            // 4 groups x 64 dims
  float accd = 0.f;
  for (int s = g; s < SEQ; s += 4)
    accd += sprob[s] * vg[(size_t)s * DM + h * DHEAD + d];
  part[g][d] = accd;
  __syncthreads();
  if (tid < DHEAD) {
    float og = (part[0][tid] + part[1][tid] + part[2][tid] + part[3][tid]) / tot;
    attb[h * DHEAD + tid] = (bf16)og;        // row s=0
  }
}

// ---------------------------------------------------------------- classifier head

__global__ void head_kernel(const float* __restrict__ x,
                            const float* __restrict__ hg, const float* __restrict__ hb,
                            const float* __restrict__ W1, const float* __restrict__ b1,
                            const float* __restrict__ W2, const float* __restrict__ b2,
                            const float* __restrict__ Wout, const float* __restrict__ bout,
                            float* __restrict__ out) {
  __shared__ float hbuf[DM];
  __shared__ float h2[DM];
  __shared__ float red[256];
  int tid = threadIdx.x;

  float e[3]; float ls = 0.f;
#pragma unroll
  for (int i = 0; i < 3; ++i) { int d = tid + i * 256; e[i] = x[d]; ls += e[i]; }
  float mean = block_sum256(ls, red) * (1.0f / DM);
  float lv = 0.f;
#pragma unroll
  for (int i = 0; i < 3; ++i) { float t = e[i] - mean; lv += t * t; }
  float rstd = rsqrtf(block_sum256(lv, red) * (1.0f / DM) + 1e-5f);
#pragma unroll
  for (int i = 0; i < 3; ++i) {
    int d = tid + i * 256;
    hbuf[d] = (e[i] - mean) * rstd * hg[d] + hb[d];
  }
  __syncthreads();

  for (int j = tid; j < DM; j += 256) {
    float a = b1[j];
    for (int k = 0; k < DM; ++k) a += hbuf[k] * W1[(size_t)k * DM + j];
    h2[j] = gelu_f(a);
  }
  __syncthreads();
  for (int j = tid; j < DM; j += 256) {
    float a = b2[j];
    for (int k = 0; k < DM; ++k) a += h2[k] * W2[(size_t)k * DM + j];
    hbuf[j] = gelu_f(a);   // reuse hbuf (phase 1 consumers done)
  }
  __syncthreads();

  float lacc = 0.f;
  for (int k = tid; k < DM; k += 256) lacc += hbuf[k] * Wout[k];
  float logit = block_sum256(lacc, red) + bout[0];
  if (tid == 0) out[0] = 1.0f / (1.0f + expf(-logit));
}

// ---------------------------------------------------------------- launch

extern "C" void kernel_launch(void* const* d_in, const int* in_sizes, int n_in,
                              void* d_out, int out_size, void* d_ws, size_t ws_size,
                              hipStream_t stream) {
  const int*   ids  = (const int*)d_in[0];
  const int*   am   = (const int*)d_in[1];
  const float* wemb = (const float*)d_in[2];
  const float* pemb = (const float*)d_in[3];
  const float* embg = (const float*)d_in[4];
  const float* embb = (const float*)d_in[5];
  const float* Wq  = (const float*)d_in[6];  const float* bq  = (const float*)d_in[7];
  const float* Wk  = (const float*)d_in[8];  const float* bk  = (const float*)d_in[9];
  const float* Wv  = (const float*)d_in[10]; const float* bv  = (const float*)d_in[11];
  const float* Wqg = (const float*)d_in[12]; const float* bqg = (const float*)d_in[13];
  const float* Wkg = (const float*)d_in[14]; const float* bkg = (const float*)d_in[15];
  const float* Wvg = (const float*)d_in[16]; const float* bvg = (const float*)d_in[17];
  const float* Wo  = (const float*)d_in[18]; const float* bo  = (const float*)d_in[19];
  const float* g1  = (const float*)d_in[20]; const float* b1  = (const float*)d_in[21];
  const float* Wf1 = (const float*)d_in[22]; const float* bf1 = (const float*)d_in[23];
  const float* Wf2 = (const float*)d_in[24]; const float* bf2 = (const float*)d_in[25];
  const float* g2  = (const float*)d_in[26]; const float* b2  = (const float*)d_in[27];
  const float* hng = (const float*)d_in[28]; const float* hnb = (const float*)d_in[29];
  const float* W1  = (const float*)d_in[30]; const float* b1h = (const float*)d_in[31];
  const float* W2  = (const float*)d_in[32]; const float* b2h = (const float*)d_in[33];
  const float* Wou = (const float*)d_in[34]; const float* bou = (const float*)d_in[35];

  char* wp = (char*)d_ws;
  auto carve = [&](size_t bytes) -> void* {
    void* p = (void*)wp;
    wp += (bytes + 255) & ~(size_t)255;
    return p;
  };
  float* x    = (float*)carve((size_t)SEQ * DM * 4);
  bf16*  xb   = (bf16*) carve((size_t)SEQ * DM * 2);
  float* tmp  = (float*)carve((size_t)SEQ * FFD * 4);
  bf16*  qbuf = (bf16*) carve((size_t)SEQ * DM * 2);
  bf16*  kbuf = (bf16*) carve((size_t)SEQ * DM * 2);
  bf16*  vbuf = (bf16*) carve((size_t)SEQ * DM * 2);
  float* kg   = (float*)carve((size_t)SEQ * DM * 4);
  float* vg   = (float*)carve((size_t)SEQ * DM * 4);
  bf16*  attb = (bf16*) carve((size_t)SEQ * DM * 2);
  bf16*  h1b  = (bf16*) carve((size_t)SEQ * FFD * 2);
  float* scr  = (float*)carve((size_t)NHEAD * SEQ * BAND * 4);
  float* sgb  = (float*)carve((size_t)NHEAD * SEQ * 4);
  float* pgb  = (float*)carve((size_t)NHEAD * SEQ * 4);
  bf16*  pbb  = (bf16*) carve((size_t)NHEAD * SEQ * BAND * 2);
  float* qg0  = (float*)carve((size_t)DM * 4);

  dim3 blk(256);
  dim3 gN768(DM / 64, SEQ / 128);    // (12,16)
  dim3 gN3072(FFD / 64, SEQ / 128);  // (48,16)
  dim3 gAttn(NCH, NHEAD);            // (16,12)

  embed_ln_kernel<<<SEQ, blk, 0, stream>>>(ids, wemb, pemb, embg, embb, x, xb);

  for (int l = 0; l < NLAYER; ++l) {
    size_t oDD = (size_t)l * DM * DM, oD = (size_t)l * DM;
    size_t oDF = (size_t)l * DM * FFD, oF = (size_t)l * FFD;

    gemm_wmma_kernel<<<gN768, blk, 0, stream>>>(xb, Wq + oDD, bq + oD, nullptr, qbuf,
                                                SEQ, DM, DM, 0, 0.125f);
    gemm_wmma_kernel<<<gN768, blk, 0, stream>>>(xb, Wk + oDD, bk + oD, nullptr, kbuf,
                                                SEQ, DM, DM, 0, 1.0f);
    gemm_wmma_kernel<<<gN768, blk, 0, stream>>>(xb, Wv + oDD, bv + oD, nullptr, vbuf,
                                                SEQ, DM, DM, 0, 1.0f);
    gemm_wmma_kernel<<<gN768, blk, 0, stream>>>(xb, Wkg + oDD, bkg + oD, kg, nullptr,
                                                SEQ, DM, DM, 0, 1.0f);
    gemm_wmma_kernel<<<gN768, blk, 0, stream>>>(xb, Wvg + oDD, bvg + oD, vg, nullptr,
                                                SEQ, DM, DM, 0, 1.0f);

    global_col_kernel<<<(NHEAD * SEQ) / 256, blk, 0, stream>>>(qbuf, kbuf, am, sgb);
    attn_scores_kernel<<<gAttn, blk, 0, stream>>>(qbuf, kbuf, am, scr);
    attn_softmax_kernel<<<(NHEAD * SEQ) / 8, blk, 0, stream>>>(scr, sgb, pgb, pbb);
    attn_out_kernel<<<gAttn, blk, 0, stream>>>(pbb, vbuf, pgb, attb);

    matvec_row0_kernel<<<DM / 256, blk, 0, stream>>>(x, Wqg + oDD, bqg + oD, qg0, 0.125f);
    global_row_kernel<<<NHEAD, blk, 0, stream>>>(qg0, kg, vg, am, attb);

    gemm_wmma_kernel<<<gN768, blk, 0, stream>>>(attb, Wo + oDD, bo + oD, tmp, nullptr,
                                                SEQ, DM, DM, 0, 1.0f);
    add_ln_kernel<<<SEQ, blk, 0, stream>>>(x, tmp, g1 + oD, b1 + oD, xb);

    gemm_wmma_kernel<<<gN3072, blk, 0, stream>>>(xb, Wf1 + oDF, bf1 + oF, nullptr, h1b,
                                                 SEQ, FFD, DM, 1, 1.0f);
    gemm_wmma_kernel<<<gN768, blk, 0, stream>>>(h1b, Wf2 + oDF, bf2 + oD, tmp, nullptr,
                                                SEQ, DM, FFD, 0, 1.0f);
    add_ln_kernel<<<SEQ, blk, 0, stream>>>(x, tmp, g2 + oD, b2 + oD, xb);
  }

  head_kernel<<<1, blk, 0, stream>>>(x, hng, hnb, W1, b1h, W2, b2h, Wou, bou,
                                     (float*)d_out);
}
